// Net_78357383348450
// MI455X (gfx1250) — matrix operand order repaired and checked by reading, not compiled, not run on hardware
//
#include <hip/hip_runtime.h>
#include <hip/hip_bf16.h>

typedef __attribute__((ext_vector_type(16))) __bf16 v16bf;
typedef __attribute__((ext_vector_type(8)))  __bf16 v8bf;
typedef __attribute__((ext_vector_type(8)))  float  v8f;
typedef __attribute__((__vector_size__(16))) int    vi4;

static __device__ __forceinline__ __bf16 f2bf(float f) { return (__bf16)f; }

static __device__ __forceinline__ v8f wmma_bf16(v16bf a, v16bf b, v8f c) {
  return __builtin_amdgcn_wmma_f32_16x16x32_bf16(false, a, false, b, (short)0, c,
                                                 false, false);
}

static __device__ __forceinline__ v16bf cat8(v8bf lo, v8bf hi) {
  return __builtin_shufflevector(lo, hi, 0, 1, 2, 3, 4, 5, 6, 7,
                                         8, 9, 10, 11, 12, 13, 14, 15);
}

// ---- CDNA5 async global->LDS copy (16 B per lane), with safe fallback ------
static __device__ __forceinline__ void async_copy16(const __bf16* g, __bf16* l) {
#if __has_builtin(__builtin_amdgcn_global_load_async_to_lds_b128)
  // Signature (from hipcc diagnostic): (v4i AS(1)*, v4i AS(3)*, imm offset, imm cpol)
  __builtin_amdgcn_global_load_async_to_lds_b128(
      (__attribute__((address_space(1))) vi4*)g,
      (__attribute__((address_space(3))) vi4*)l, 0, 0);
#else
  *(uint4*)l = *(const uint4*)g;
#endif
}

static __device__ __forceinline__ void wait_async() {
#if __has_builtin(__builtin_amdgcn_s_wait_asynccnt)
  __builtin_amdgcn_s_wait_asynccnt(0);
#elif __has_builtin(__builtin_amdgcn_global_load_async_to_lds_b128)
  asm volatile("s_wait_asynccnt 0x0" ::: "memory");
#endif
}

// ---------------------------------------------------------------------------
// Kernel 0: W [512,256] f32 -> Wt [256,512] bf16 (transposed);
//           means [1000,256] f32 -> meansb [1008,256] bf16 (padded rows = 0)
// ---------------------------------------------------------------------------
__global__ void prep_kernel(const float* __restrict__ W,
                            const float* __restrict__ means,
                            __bf16* __restrict__ Wt,
                            __bf16* __restrict__ meansb) {
  int idx = blockIdx.x * blockDim.x + threadIdx.x;
  const int wtN = 256 * 512;
  const int mbN = 1008 * 256;
  if (idx < wtN) {
    int n = idx >> 9;
    int k = idx & 511;
    Wt[idx] = f2bf(W[k * 256 + n]);
  } else if (idx < wtN + mbN) {
    int j = idx - wtN;
    int r = j >> 8;
    int c = j & 255;
    meansb[j] = f2bf(r < 1000 ? means[r * 256 + c] : 0.0f);
  }
}

// A-fragment raw storage: 16 floats kept as 4 float4s (guaranteed VGPRs).
struct AReg { float4 a, b, c, d; };

static __device__ __forceinline__ void loadA(const float* xrow, int ks, int half,
                                             AReg& r) {
  int ka = ks * 32 + half * 8;     // 16-bit A layout: K ka..+7 and ka+16..+23
  r.a = *(const float4*)(xrow + ka);
  r.b = *(const float4*)(xrow + ka + 4);
  r.c = *(const float4*)(xrow + ka + 16);
  r.d = *(const float4*)(xrow + ka + 20);
}

static __device__ __forceinline__ v16bf cvtA(const AReg& r) {
  v16bf a;
  a[0]  = f2bf(r.a.x); a[1]  = f2bf(r.a.y); a[2]  = f2bf(r.a.z); a[3]  = f2bf(r.a.w);
  a[4]  = f2bf(r.b.x); a[5]  = f2bf(r.b.y); a[6]  = f2bf(r.b.z); a[7]  = f2bf(r.b.w);
  a[8]  = f2bf(r.c.x); a[9]  = f2bf(r.c.y); a[10] = f2bf(r.c.z); a[11] = f2bf(r.c.w);
  a[12] = f2bf(r.d.x); a[13] = f2bf(r.d.y); a[14] = f2bf(r.d.z); a[15] = f2bf(r.d.w);
  return a;
}

// ---------------------------------------------------------------------------
// Kernel 1: feats = x @ W ; L2-normalize rows ; store preds bf16.
// 8 waves/block; Wt k-slices staged in LDS (double-buffered, async).
// LDS slice layout: 256 rows (n) x 32 K, row stride 40 elems (+16 B pad).
// ---------------------------------------------------------------------------
#define K1_STRIDE 40
__global__ void __launch_bounds__(256)
gemm1_norm_kernel(const float* __restrict__ x,
                  const __bf16* __restrict__ Wt,
                  __bf16* __restrict__ preds) {
  __shared__ __align__(16) __bf16 sW[2][256 * K1_STRIDE];

  const int tid   = threadIdx.x;
  const int lane  = tid & 31;
  const int strip = blockIdx.x * 8 + (tid >> 5);
  const int row16 = strip * 16;
  const int half  = lane >> 4;
  const int lid   = lane & 15;

  v8f acc[16];
#pragma unroll
  for (int t = 0; t < 16; ++t) acc[t] = (v8f){};

  const float* xrow = x + (size_t)(row16 + lid) * 512;

  // cooperative slice loader: 1024 x 16 B chunks, 4 per thread
  auto load_slice = [&](int ks, int buf) {
#pragma unroll
    for (int i = 0; i < 4; ++i) {
      int c  = tid + i * 256;
      int n  = c >> 2;
      int ko = (c & 3) * 8;
      async_copy16(Wt + (size_t)n * 512 + ks * 32 + ko, &sW[buf][n * K1_STRIDE + ko]);
    }
  };

  auto compute = [&](int buf, const AReg& ar) {
    v16bf afrag = cvtA(ar);
    const __bf16* sbase = &sW[buf][0];
#pragma unroll
    for (int t = 0; t < 16; ++t) {
      const __bf16* wrow = sbase + (t * 16 + lid) * K1_STRIDE + half * 8;
      v16bf bfrag = cat8(*(const v8bf*)(wrow), *(const v8bf*)(wrow + 16));
      acc[t] = wmma_bf16(afrag, bfrag, acc[t]);
    }
  };

  AReg xa, xb;
  loadA(xrow, 0, half, xa);
  load_slice(0, 0);

  for (int kp = 0; kp < 8; ++kp) {
    const int ks1 = 2 * kp + 1;
    wait_async();
    __syncthreads();
    load_slice(ks1, 1);                    // prefetch odd slice
    loadA(xrow, ks1, half, xb);
    compute(0, xa);                        // even k-step

    wait_async();
    __syncthreads();
    if (ks1 + 1 < 16) {
      load_slice(ks1 + 1, 0);              // prefetch next even slice
      loadA(xrow, ks1 + 1, half, xa);
    }
    compute(1, xb);                        // odd k-step
  }

  // Row norms: lane holds rows m = v + 8*half, column n = t*16 + lid.
  float rs[8];
#pragma unroll
  for (int v = 0; v < 8; ++v) {
    float s = 0.0f;
#pragma unroll
    for (int t = 0; t < 16; ++t) s += acc[t][v] * acc[t][v];
#pragma unroll
    for (int m = 1; m <= 8; m <<= 1) s += __shfl_xor(s, m, 32);
    rs[v] = 1.0f / fmaxf(sqrtf(s), 1e-12f);
  }

#pragma unroll
  for (int v = 0; v < 8; ++v) {
    int row = row16 + v + 8 * half;
    __bf16* prow = preds + (size_t)row * 256;
#pragma unroll
    for (int t = 0; t < 16; ++t) prow[t * 16 + lid] = f2bf(acc[t][v] * rs[v]);
  }
}

// ---------------------------------------------------------------------------
// Kernel 2: scores = preds @ means^T ; argmax (first on ties) ; one-hot.
// 8 waves/block; means class tiles staged in LDS (double-buffered, async).
// LDS tile layout: 16 rows x 256 K, row stride 264 elems (+16 B pad).
// ---------------------------------------------------------------------------
#define K2_STRIDE 264
__global__ void __launch_bounds__(256)
gemm2_argmax_kernel(const __bf16* __restrict__ preds,
                    const __bf16* __restrict__ meansb,
                    float* __restrict__ out) {
  __shared__ __align__(16) __bf16 sM[2][16 * K2_STRIDE];

  const int tid   = threadIdx.x;
  const int lane  = tid & 31;
  const int strip = blockIdx.x * 8 + (tid >> 5);
  const int row16 = strip * 16;
  const int half  = lane >> 4;
  const int lid   = lane & 15;

  // Preload 8 A fragments (preds rows, K=256), reused for all 63 class tiles.
  v16bf afrag[8];
  const __bf16* prow = preds + (size_t)(row16 + lid) * 256;
#pragma unroll
  for (int ks = 0; ks < 8; ++ks) {
    int ka = ks * 32 + half * 8;
    afrag[ks] = cat8(*(const v8bf*)(prow + ka), *(const v8bf*)(prow + ka + 16));
  }

  auto load_tile = [&](int nt, int buf) {
#pragma unroll
    for (int i = 0; i < 2; ++i) {
      int c  = tid + i * 256;        // 0..511 chunks of 16 B
      int r  = c >> 5;               // class within tile
      int ko = (c & 31) * 8;         // element offset
      async_copy16(meansb + (size_t)(nt * 16 + r) * 256 + ko,
                   &sM[buf][r * K2_STRIDE + ko]);
    }
  };

  float bv[8];
  int   bi[8];
#pragma unroll
  for (int v = 0; v < 8; ++v) { bv[v] = -__builtin_inff(); bi[v] = 0x7fffffff; }

  load_tile(0, 0);
  for (int nt = 0; nt < 63; ++nt) {
    wait_async();
    __syncthreads();
    if (nt + 1 < 63) load_tile(nt + 1, (nt + 1) & 1);

    const __bf16* tb = &sM[nt & 1][0];
    v8f acc = (v8f){};
#pragma unroll
    for (int ks = 0; ks < 8; ++ks) {
      const __bf16* mrow = tb + lid * K2_STRIDE + half * 8 + ks * 32;
      v16bf bfrag = cat8(*(const v8bf*)(mrow), *(const v8bf*)(mrow + 16));
      acc = wmma_bf16(afrag[ks], bfrag, acc);
    }

    int n = nt * 16 + lid;
#pragma unroll
    for (int v = 0; v < 8; ++v) {
      float val = (n < 1000) ? acc[v] : -__builtin_inff();
      if (val > bv[v]) { bv[v] = val; bi[v] = n; }   // strict > => first index
    }
  }

  // Cross-lane argmax within each 16-lane half (smaller index wins ties).
#pragma unroll
  for (int v = 0; v < 8; ++v) {
#pragma unroll
    for (int m = 1; m <= 8; m <<= 1) {
      float ov = __shfl_xor(bv[v], m, 32);
      int   oi = __shfl_xor(bi[v], m, 32);
      if (ov > bv[v] || (ov == bv[v] && oi < bi[v])) { bv[v] = ov; bi[v] = oi; }
    }
  }

  // Write one-hot rows (250 float4 per row, whole wave cooperates).
#pragma unroll
  for (int v = 0; v < 8; ++v) {
    int am0 = __shfl(bi[v], 0, 32);
    int am1 = __shfl(bi[v], 16, 32);
#pragma unroll
    for (int r = 0; r < 2; ++r) {
      int row = row16 + v + 8 * r;
      int am  = r ? am1 : am0;
      float* p = out + (size_t)row * 1000;
#pragma unroll
      for (int i = 0; i < 8; ++i) {
        int c4 = i * 32 + lane;
        if (c4 < 250) {
          int col = c4 * 4;
          float4 w;
          w.x = (col + 0 == am) ? 1.0f : 0.0f;
          w.y = (col + 1 == am) ? 1.0f : 0.0f;
          w.z = (col + 2 == am) ? 1.0f : 0.0f;
          w.w = (col + 3 == am) ? 1.0f : 0.0f;
          *(float4*)(p + col) = w;
        }
      }
    }
  }
}

extern "C" void kernel_launch(void* const* d_in, const int* in_sizes, int n_in,
                              void* d_out, int out_size, void* d_ws, size_t ws_size,
                              hipStream_t stream) {
  const float* x     = (const float*)d_in[0];   // [16384, 512]
  const float* W     = (const float*)d_in[1];   // [512, 256]
  const float* means = (const float*)d_in[2];   // [1000, 256]
  float* out = (float*)d_out;                   // [16384, 1000]

  char* ws = (char*)d_ws;
  __bf16* preds  = (__bf16*)ws;                                          // 8 MB
  __bf16* Wt     = (__bf16*)(ws + 16384u * 256u * 2u);                   // 256 KB
  __bf16* meansb = (__bf16*)(ws + 16384u * 256u * 2u + 256u * 512u * 2u);

  const int prepN = 256 * 512 + 1008 * 256;
  prep_kernel<<<(prepN + 255) / 256, 256, 0, stream>>>(W, means, Wt, meansb);
  gemm1_norm_kernel<<<128, 256, 0, stream>>>(x, Wt, preds);
  gemm2_argmax_kernel<<<128, 256, 0, stream>>>(preds, meansb, out);
}
